// mGaussNorm_T_66915590472011
// MI455X (gfx1250) — matrix operand order repaired and checked by reading, not compiled, not run on hardware
//
#include <hip/hip_runtime.h>

// ---------------------------------------------------------------------------
// mGaussNorm for MI455X (gfx1250, wave32, WMMA + TDM)
//   conv1 (loc):  GEMM M=16(f) x N=128(co) x K=3072(s,ci)  per batch, f16 WMMA
//   conv2 (scale):GEMM M=136->9x16 tiles x N=128 x K=9216, f16 WMMA; B stream
//                 staged via TENSOR_LOAD_TO_LDS double-buffer (TENSORcnt sync)
//   eigh:         32768 x (16x16 symmetric Jacobi) in LDS, 2 matrices / wave32
// ---------------------------------------------------------------------------

typedef __attribute__((ext_vector_type(16))) _Float16 v16h;
typedef __attribute__((ext_vector_type(8)))  float    v8f;
typedef unsigned int v4u __attribute__((ext_vector_type(4)));
typedef int          v4i __attribute__((ext_vector_type(4)));
typedef int          v8i __attribute__((ext_vector_type(8)));

#define BATCH 256
#define SEQ   24
#define CIN   128
#define COUT  128
#define FEAT  16
#define KH    8
#define KW    9
#define NPACK 136            // 17*8 = number of upper-tri entries of 16x16
#define K1    (SEQ*CIN)      // 3072
#define T1    (K1/32)        // 96 k-steps
#define K2    (CIN*KH*KW)    // 9216
#define T2    (K2/32)        // 288 k-steps
#define CHUNK_T 16           // k-steps per TDM chunk (16 KB)
#define NCHUNK  (T2/CHUNK_T) // 18

// workspace layout (bytes)
#define WS_X16   0u                         // 12,582,912 halfs = 25,165,824 B
#define WS_B1    25165824u                  //    393,216 halfs =    786,432 B
#define WS_B2    25952256u                  //  1,179,648 halfs =  2,359,296 B
#define WS_SCALE 28311552u                  //  4,456,448 f32   = 17,825,792 B
// total = 46,137,344 B

__device__ __forceinline__ float softplus_f(float x) {
    return x > 20.f ? x : log1pf(expf(x));
}

__device__ __forceinline__ v8f wmma_f16(v16h a, v16h b, v8f c) {
    // (neg_a, A, neg_b, B, c_mod, C, reuse_a, reuse_b)
    return __builtin_amdgcn_wmma_f32_16x16x32_f16(false, a, false, b,
                                                  (short)0, c, false, false);
}

// 1-D TDM copy: ndw dwords from global gptr -> LDS lptr (tracked by TENSORcnt).
// D# per CDNA5 ISA ch.8: group0 = count/lds_addr/global_addr/type,
// group1 = data_size(4B) + tensor/tile dim0 (1 row of ndw dwords).
__device__ __forceinline__ void tdm_load_1d(const void* gptr, void* lptr,
                                            unsigned ndw) {
    unsigned long long ga = (unsigned long long)(uintptr_t)gptr;
    unsigned la = (unsigned)(uintptr_t)lptr;   // low 32b of generic = LDS offset
    v4u g0;
    g0[0] = 1u;                                         // count=1 (valid, user)
    g0[1] = la;                                         // lds_addr [63:32]
    g0[2] = (unsigned)ga;                               // global_addr[31:0]
    g0[3] = (unsigned)((ga >> 32) & 0x1FFFFFFu)         // global_addr[56:32]
          | (2u << 30);                                 // type=2 ("image")
    v8i g1;
    g1[0] = 0x00020000;                        // data_size=4B; no multicast
    g1[1] = (int)((ndw & 0xFFFFu) << 16);      // tensor_dim0[15:0]  (bits 63:48)
    g1[2] = (int)((ndw >> 16) | (1u << 16));   // tensor_dim0[31:16] | tensor_dim1=1
    g1[3] = (int)((ndw & 0xFFFFu) << 16);      // tile_dim0 (bits 127:112)
    g1[4] = 1;                                 // tile_dim1=1, tile_dim2=0
    g1[5] = (int)ndw;                          // tensor_dim0_stride[31:0]
    g1[6] = 0;
    g1[7] = 0;
    v4i g2 = {0, 0, 0, 0};                     // <=2-D tensor: groups 2/3 unused
    v4i g3 = {0, 0, 0, 0};
#if __clang_major__ >= 23                      // amdgpu-toolchain (therock HIP)
    v8i g4 = {0, 0, 0, 0, 0, 0, 0, 0};
    __builtin_amdgcn_tensor_load_to_lds(g0, g1, g2, g3, g4, 0);
#else                                          // ROCm 7.2 / clang-22: 5-arg form
    __builtin_amdgcn_tensor_load_to_lds(g0, g1, g2, g3, 0);
#endif
}

// ---------------------------------------------------------------- precompute
__global__ __launch_bounds__(256) void cvt_x_f16(const float* __restrict__ x,
                                                 _Float16* __restrict__ x16) {
    int i = blockIdx.x * 256 + threadIdx.x;        // 12,582,912 elements
    x16[i] = (_Float16)x[i];
}

// Pack n_w into B-fragment layout: [cot][t][lane][j] ; K order k = s*128+ci
__global__ __launch_bounds__(256) void pack_b1(const float* __restrict__ n_w,
                                               _Float16* __restrict__ B1) {
    int idx  = blockIdx.x * 256 + threadIdx.x;     // 393,216 elements
    int j    = idx & 15;
    int u    = idx >> 4;
    int lane = u & 31; u >>= 5;
    int t    = u % T1;
    int cot  = u / T1;
    int k    = t * 32 + ((j & 8) ? 16 : 0) + ((lane & 16) ? 8 : 0) + (j & 7);   // lanes 0-15: K 0-7,16-23; 16-31: K 8-15,24-31
    int s    = k >> 7;
    int ci   = k & 127;
    int co   = cot * 16 + (lane & 15);
    B1[idx]  = (_Float16)n_w[(co * CIN + ci) * SEQ + s];
}

// Pack p_w into B-fragment layout: [cot][t][lane][j] ; K order k = ci*72 + kh*9 + kw
__global__ __launch_bounds__(256) void pack_b2(const float* __restrict__ p_w,
                                               _Float16* __restrict__ B2) {
    int idx  = blockIdx.x * 256 + threadIdx.x;     // 1,179,648 elements
    int j    = idx & 15;
    int u    = idx >> 4;
    int lane = u & 31; u >>= 5;
    int t    = u % T2;
    int cot  = u / T2;
    int k    = t * 32 + ((j & 8) ? 16 : 0) + ((lane & 16) ? 8 : 0) + (j & 7);
    int ci   = k / 72;
    int r    = k - ci * 72;                        // kh*9 + kw
    int co   = cot * 16 + (lane & 15);
    B2[idx]  = (_Float16)p_w[(co * CIN + ci) * (KH * KW) + r];
}

// ----------------------------------------------------------- conv1 (loc) GEMM
// one wave per (b, cot): D[f, co] = sum_k A[f,k] * B[k,co]; 96 WMMA k-steps
__global__ __launch_bounds__(256) void conv1_wmma(const _Float16* __restrict__ x16,
                                                  const _Float16* __restrict__ B1,
                                                  const float* __restrict__ n_b,
                                                  float* __restrict__ loc) {
    const int lane = threadIdx.x & 31;
    const int wid  = blockIdx.x * 8 + (threadIdx.x >> 5);   // 2048 waves
    const int b    = wid >> 3;
    const int cot  = wid & 7;
    const int m    = lane & 15;                 // A row = f
    const int kbA  = (lane & 16) ? 8 : 0;       // A lane-half K offset
    const _Float16* xb = x16 + b * (SEQ * CIN * FEAT);
    const _Float16* bp = B1 + (size_t)cot * T1 * 512 + lane * 16;

    v8f acc = {0.f, 0.f, 0.f, 0.f, 0.f, 0.f, 0.f, 0.f};
    for (int t = 0; t < T1; ++t) {
        v16h a;
#pragma unroll
        for (int v = 0; v < 8; ++v) {
            int k0 = t * 32 + ((v & 4) ? 16 : 0) + kbA + (v & 3) * 2;
            int s0 = k0 >> 7, c0 = k0 & 127;
            int k1 = k0 + 1;
            int s1 = k1 >> 7, c1 = k1 & 127;
            a[2 * v]     = xb[(s0 * CIN + c0) * FEAT + m];
            a[2 * v + 1] = xb[(s1 * CIN + c1) * FEAT + m];
        }
        v16h bf = *(const v16h*)bp;
        __builtin_prefetch(bp + 512, 0, 0);     // global_prefetch_b8 on next frag
        acc = wmma_f16(a, bf, acc);
        bp += 512;
    }
    // D layout: vgpr r, lanes 0-15 -> (M=r, N=lane); lanes 16-31 -> (M=8+r, N=lane-16)
    const int   co = cot * 16 + (lane & 15);
    const int   f0 = (lane & 16) ? 8 : 0;
    const float bias = n_b[co];
    float* op = loc + (size_t)b * (COUT * FEAT) + co * FEAT + f0;
    float4 r0, r1;
    r0.x = softplus_f(acc[0] + bias); r0.y = softplus_f(acc[1] + bias);
    r0.z = softplus_f(acc[2] + bias); r0.w = softplus_f(acc[3] + bias);
    r1.x = softplus_f(acc[4] + bias); r1.y = softplus_f(acc[5] + bias);
    r1.z = softplus_f(acc[6] + bias); r1.w = softplus_f(acc[7] + bias);
    *(float4*)op       = r0;
    *(float4*)(op + 4) = r1;
}

// --------------------------------------------------------- conv2 (scale) GEMM
// one block per (b, cot): 9 waves, one per 16-row M tile (p = h*8+w, 136 rows).
// All 9 waves share one B-fragment stream -> staged through LDS by the Tensor
// Data Mover, double-buffered, synchronized with s_wait_tensorcnt + barrier.
__global__ __launch_bounds__(288) void conv2_wmma(const _Float16* __restrict__ x16,
                                                  const _Float16* __restrict__ B2,
                                                  const float* __restrict__ p_b,
                                                  float* __restrict__ sc) {
    __shared__ _Float16 ldsB[2][CHUNK_T * 512];   // 2 x 16 KB
    const int tid  = threadIdx.x;
    const int lane = tid & 31;
    const int pt   = tid >> 5;                  // wave id = M tile, 0..8
    const int b    = blockIdx.x >> 3;
    const int cot  = blockIdx.x & 7;
    const int m    = lane & 15;
    const int prow = pt * 16 + m;
    const bool valid = prow < NPACK;
    const int h    = prow >> 3;                 // 0..16 (garbage rows masked)
    const int w    = prow & 7;
    const int kbA  = (lane & 16) ? 8 : 0;
    const _Float16* xb = x16 + (size_t)b * (SEQ * CIN * FEAT);
    const _Float16* bg = B2 + (size_t)cot * T2 * 512;   // block's B stream

    if (tid < 32)                                       // wave 0 drives the TDM
        tdm_load_1d(bg, &ldsB[0][0], CHUNK_T * 512 / 2);

    v8f acc = {0.f, 0.f, 0.f, 0.f, 0.f, 0.f, 0.f, 0.f};
    for (int c = 0; c < NCHUNK; ++c) {
        if (tid < 32) __builtin_amdgcn_s_wait_tensorcnt(0);
        __syncthreads();                                 // buf[c&1] ready
        if (tid < 32 && c + 1 < NCHUNK)                  // overlap next chunk
            tdm_load_1d(bg + (size_t)(c + 1) * CHUNK_T * 512,
                        &ldsB[(c + 1) & 1][0], CHUNK_T * 512 / 2);
        const _Float16* lb = &ldsB[c & 1][lane * 16];
        for (int tt = 0; tt < CHUNK_T; ++tt) {
            const int t = c * CHUNK_T + tt;
            v16h a;
#pragma unroll
            for (int v = 0; v < 8; ++v) {
                int k0  = t * 32 + ((v & 4) ? 16 : 0) + kbA + (v & 3) * 2;
                int ci0 = k0 / 72;  int r0 = k0 - ci0 * 72;
                int kh0 = r0 / 9;   int kw0 = r0 - kh0 * 9;
                int k1  = k0 + 1;
                int ci1 = k1 / 72;  int r1 = k1 - ci1 * 72;
                int kh1 = r1 / 9;   int kw1 = r1 - kh1 * 9;
                a[2 * v]     = valid ? xb[((h + kh0) * CIN + ci0) * FEAT + (w + kw0)]
                                     : (_Float16)0.f;
                a[2 * v + 1] = valid ? xb[((h + kh1) * CIN + ci1) * FEAT + (w + kw1)]
                                     : (_Float16)0.f;
            }
            v16h bf = *(const v16h*)(lb + tt * 512);     // ds_load from staged buf
            acc = wmma_f16(a, bf, acc);
        }
        __syncthreads();                                 // done reading buf[c&1]
    }

    const int   co = cot * 16 + (lane & 15);
    const int   p0 = pt * 16 + ((lane & 16) ? 8 : 0);
    const float bias = p_b[co];
    float* op = sc + ((size_t)b * COUT + co) * NPACK;
#pragma unroll
    for (int r = 0; r < 8; ++r) {
        int p = p0 + r;
        if (p < NPACK) op[p] = softplus_f(acc[r] + bias);   // softplus fused here
    }
}

// -------------------------------------------------- eigh + reconstruct (LDS)
// 16 matrices per 256-thread block (2 per wave32); cyclic Jacobi, 8 sweeps
__global__ __launch_bounds__(256) void eigh16(const float* __restrict__ sc,
                                              float* __restrict__ pd) {
    __shared__ float As[16][16][17];   // +1 pad vs 64-bank LDS
    __shared__ float Vs[16][16][17];
    __shared__ float Ev[16][16];
    const int tid = threadIdx.x;
    const int mm  = tid >> 4;          // matrix slot in block
    const int i   = tid & 15;          // row owned by this lane
    const int mat = blockIdx.x * 16 + mm;          // = b*128 + co
    const float* s = sc + (size_t)mat * NPACK;

    // build symmetric 16x16 from row-major upper triangle (softplus already applied)
    for (int j = 0; j < 16; ++j) {
        int lo = i < j ? i : j;
        int hi = i < j ? j : i;
        int up = lo * 16 - (lo * (lo - 1)) / 2 + (hi - lo);
        As[mm][i][j] = s[up];
        Vs[mm][i][j] = (i == j) ? 1.f : 0.f;
    }
    __syncthreads();

    for (int sweep = 0; sweep < 8; ++sweep) {
        for (int p = 0; p < 15; ++p) {
            for (int q = p + 1; q < 16; ++q) {
                float app = As[mm][p][p];
                float aqq = As[mm][q][q];
                float apq = As[mm][p][q];
                float c = 1.f, sn = 0.f;
                if (fabsf(apq) > 1e-12f) {
                    float tau = (aqq - app) / (2.f * apq);
                    float tt  = copysignf(1.f, tau) /
                                (fabsf(tau) + sqrtf(1.f + tau * tau));
                    c  = 1.f / sqrtf(1.f + tt * tt);
                    sn = tt * c;
                }
                __syncthreads();                   // all params read before writes
                // A <- A*J and V <- V*J : lane i rotates its row's (p,q) entries
                float aip = As[mm][i][p], aiq = As[mm][i][q];
                As[mm][i][p] = c * aip - sn * aiq;
                As[mm][i][q] = sn * aip + c * aiq;
                float vip = Vs[mm][i][p], viq = Vs[mm][i][q];
                Vs[mm][i][p] = c * vip - sn * viq;
                Vs[mm][i][q] = sn * vip + c * viq;
                __syncthreads();
                // A <- J^T*A : lane i rotates column i of rows p,q
                float api = As[mm][p][i], aqi = As[mm][q][i];
                As[mm][p][i] = c * api - sn * aqi;
                As[mm][q][i] = sn * api + c * aqi;
                __syncthreads();
            }
        }
    }

    // clamp eigenvalues, reconstruct pd = V * diag(cl) * V^T  (symmetric by build)
    Ev[mm][i] = fmaxf(As[mm][i][i], 1e-5f);
    __syncthreads();
    float* op = pd + (size_t)mat * 256 + i * 16;
    for (int j = 0; j < 16; ++j) {
        float sum = 0.f;
#pragma unroll
        for (int k = 0; k < 16; ++k)
            sum = fmaf(Vs[mm][i][k] * Ev[mm][k], Vs[mm][j][k], sum);
        op[j] = sum;
    }
}

// ---------------------------------------------------------------------------
extern "C" void kernel_launch(void* const* d_in, const int* in_sizes, int n_in,
                              void* d_out, int out_size, void* d_ws, size_t ws_size,
                              hipStream_t stream) {
    (void)in_sizes; (void)n_in; (void)out_size; (void)ws_size;
    const float* x   = (const float*)d_in[0];   // (256,24,128,16)
    const float* n_w = (const float*)d_in[1];   // (128,128,24,1)
    const float* n_b = (const float*)d_in[2];   // (128,)
    const float* p_w = (const float*)d_in[3];   // (128,128,8,9)
    const float* p_b = (const float*)d_in[4];   // (128,)
    float* out = (float*)d_out;                 // loc[524288] ++ pd[8388608]

    char* ws = (char*)d_ws;                     // needs 46,137,344 B
    _Float16* x16 = (_Float16*)(ws + WS_X16);
    _Float16* B1  = (_Float16*)(ws + WS_B1);
    _Float16* B2  = (_Float16*)(ws + WS_B2);
    float*    sc  = (float*)   (ws + WS_SCALE);

    cvt_x_f16 <<<49152, 256, 0, stream>>>(x, x16);
    pack_b1   <<<1536,  256, 0, stream>>>(n_w, B1);
    pack_b2   <<<4608,  256, 0, stream>>>(p_w, B2);
    conv1_wmma<<<256,   256, 0, stream>>>(x16, B1, n_b, out);
    conv2_wmma<<<2048,  288, 0, stream>>>(x16, B2, p_b, sc);
    eigh16    <<<2048,  256, 0, stream>>>(sc, out + 524288);
}
